// MultilevelDetectionGenerator_52252572123573
// MI455X (gfx1250) — compile-verified, matrix operand order checked by reading, not run on hardware
//
#include <hip/hip_runtime.h>
#include <hip/hip_bf16.h>
#include <stdint.h>

// ---------------- problem constants ----------------
#define BATCH    8
#define NBOX     100000
#define NCLS     90
#define NBC      (BATCH * NCLS)      // 720 (b,c) pairs
#define NC_FLAT  (NBOX * NCLS)       // 9,000,000 scores per batch
#define KSEL     512                 // PRE_NMS_K
#define MAXDET   100
#define NBINS    1024
#define BINCAP   1024
#define SCORE_THR 0.05f
#define NEGV     (-1.0e9f)

typedef __attribute__((ext_vector_type(2)))  float    v2f;
typedef __attribute__((ext_vector_type(8)))  float    v8f;
typedef __attribute__((ext_vector_type(16))) _Float16 v16h;

#if __has_builtin(__builtin_amdgcn_wmma_f32_16x16x4_f32)
  #define AREA_SCALE 1.0f
#elif __has_builtin(__builtin_amdgcn_wmma_f32_16x16x32_f16)
  #define AREA_SCALE 0.00390625f     // 2^-8: keeps areas (<=512^2) inside f16 range
#else
  #define AREA_SCALE 1.0f
#endif

// D = A(16x4) x B(4x16): A row m = [area_m, 1, 0, 0], B col n = [1, area_n, 0, 0]^T
// => D[m][n] = area_m + area_n  (scaled by AREA_SCALE on the f16 path).
// Layouts per CDNA5 ISA 7.12.2: A: lane=M, VGPR=K (lanes16-31 -> K=2,3);
// B: lane=N, VGPR=K; D: vgpr v, lane l -> M = v + (l>=16 ? 8 : 0), N = l&15.
__device__ __forceinline__ v8f tile_pair_sums(const float* __restrict__ ar,
                                              int i0, int j0, int laneLo, int hi) {
#if __has_builtin(__builtin_amdgcn_wmma_f32_16x16x4_f32)
  v2f a, b;
  a.x = hi ? 0.0f : ar[i0 + laneLo];
  a.y = hi ? 0.0f : 1.0f;
  b.x = hi ? 0.0f : 1.0f;
  b.y = hi ? 0.0f : ar[j0 + laneLo];
  v8f c;
  #pragma unroll
  for (int q = 0; q < 8; ++q) c[q] = 0.0f;
  return __builtin_amdgcn_wmma_f32_16x16x4_f32(false, a, false, b, (short)0, c, false, false);
#elif __has_builtin(__builtin_amdgcn_wmma_f32_16x16x32_f16)
  v16h a, b;
  #pragma unroll
  for (int q = 0; q < 16; ++q) { a[q] = (_Float16)0.0f; b[q] = (_Float16)0.0f; }
  if (!hi) { // lanes 0-15: elements 0,1 (K=0,1) carry [area,1] / [1,area]
    a[0] = (_Float16)(ar[i0 + laneLo] * AREA_SCALE);
    a[1] = (_Float16)1.0f;
    b[0] = (_Float16)1.0f;
    b[1] = (_Float16)(ar[j0 + laneLo] * AREA_SCALE);
  }
  v8f c;
  #pragma unroll
  for (int q = 0; q < 8; ++q) c[q] = 0.0f;
  return __builtin_amdgcn_wmma_f32_16x16x32_f16(false, a, false, b, (short)0, c, false, false);
#else  // host-compilation pass / no WMMA: plain VALU
  v8f d;
  #pragma unroll
  for (int v = 0; v < 8; ++v) d[v] = (ar[i0 + v + hi * 8] + ar[j0 + laneLo]) * AREA_SCALE;
  return d;
#endif
}

// ---------------- pass 1: per-(b,c) score histogram ----------------
#define HITEMS 16
#define HCHUNK (256 * HITEMS)

__global__ void det_hist_kernel(const float* __restrict__ scores, int* __restrict__ hist) {
  const int b = blockIdx.y;
  const size_t base = (size_t)b * NC_FLAT;
  const int flat0 = blockIdx.x * HCHUNK + threadIdx.x;
  #pragma unroll
  for (int it = 0; it < HITEMS; ++it) {
    int flat = flat0 + it * 256;
    int pf = flat + HCHUNK; if (pf > NC_FLAT - 1) pf = NC_FLAT - 1;
    __builtin_prefetch(&scores[base + pf], 0, 0);           // global_prefetch_b8
    if (flat < NC_FLAT) {
      float s = scores[base + flat];
      if (s >= SCORE_THR) {
        int c = flat % NCLS;
        int bin = (int)(s * (float)NBINS);
        if (bin > NBINS - 1) bin = NBINS - 1;
        atomicAdd(&hist[(b * NCLS + c) * NBINS + bin], 1);
      }
    }
  }
}

// ---------------- pass 2: find cut bin so that count(bin > b*) < 512 <= count(bin >= b*)
__global__ void det_cut_kernel(const int* __restrict__ hist, int* __restrict__ cut) {
  const int bc = blockIdx.x;
  __shared__ int h[NBINS];
  for (int i = threadIdx.x; i < NBINS; i += blockDim.x) h[i] = hist[bc * NBINS + i];
  __syncthreads();
  if (threadIdx.x == 0) {
    int cum = 0, binIdx = -1, k1 = 0;
    for (int bn = NBINS - 1; bn >= 0; --bn) {
      int nc = cum + h[bn];
      if (nc >= KSEL) { binIdx = bn; k1 = cum; break; }
      cum = nc;
    }
    int need = (binIdx < 0) ? 0 : (KSEL - k1);
    cut[bc * 2 + 0] = binIdx;
    cut[bc * 2 + 1] = need;
  }
}

// ---------------- pass 3: compaction (above-cut -> cand, boundary bin -> side buf)
__global__ void det_compact_kernel(const float* __restrict__ scores, const int* __restrict__ cut,
                                   float* __restrict__ candS, int* __restrict__ candI,
                                   int* __restrict__ candCount,
                                   float* __restrict__ binS, int* __restrict__ binI,
                                   int* __restrict__ binCount) {
  const int b = blockIdx.y;
  __shared__ int binStar[NCLS];
  for (int c = threadIdx.x; c < NCLS; c += blockDim.x) binStar[c] = cut[(b * NCLS + c) * 2];
  __syncthreads();
  const size_t base = (size_t)b * NC_FLAT;
  const int flat0 = blockIdx.x * HCHUNK + threadIdx.x;
  #pragma unroll
  for (int it = 0; it < HITEMS; ++it) {
    int flat = flat0 + it * 256;
    int pf = flat + HCHUNK; if (pf > NC_FLAT - 1) pf = NC_FLAT - 1;
    __builtin_prefetch(&scores[base + pf], 0, 0);
    if (flat < NC_FLAT) {
      float s = scores[base + flat];
      if (s >= SCORE_THR) {
        int c = flat % NCLS;
        int n = flat / NCLS;
        int bin = (int)(s * (float)NBINS);
        if (bin > NBINS - 1) bin = NBINS - 1;
        int bs = binStar[c];
        int bc = b * NCLS + c;
        if (bin > bs) {
          int pos = atomicAdd(&candCount[bc], 1);           // exact count < 512 by construction
          if (pos < KSEL) { candS[bc * KSEL + pos] = s; candI[bc * KSEL + pos] = n; }
        } else if (bin == bs) {
          int pos = atomicAdd(&binCount[bc], 1);
          if (pos < BINCAP) { binS[bc * BINCAP + pos] = s; binI[bc * BINCAP + pos] = n; }
        }
      }
    }
  }
}

// ---------------- pass 4: sort boundary bin, append top `need`, pad to 512
__global__ void __launch_bounds__(512) det_binselect_kernel(
    const float* __restrict__ binS, const int* __restrict__ binI,
    const int* __restrict__ binCount, const int* __restrict__ cut,
    float* __restrict__ candS, int* __restrict__ candI, const int* __restrict__ candCount) {
  const int bc = blockIdx.x;
  __shared__ float s[BINCAP];
  __shared__ int   id[BINCAP];
  int cnt = binCount[bc]; if (cnt > BINCAP) cnt = BINCAP;
  for (int i = threadIdx.x; i < BINCAP; i += 512) {
    if (i < cnt) { s[i] = binS[bc * BINCAP + i]; id[i] = binI[bc * BINCAP + i]; }
    else         { s[i] = NEGV;                  id[i] = 0x7fffffff; }
  }
  __syncthreads();
  // bitonic sort, descending score (ties: lower index first)
  for (int k = 2; k <= BINCAP; k <<= 1) {
    for (int j = k >> 1; j > 0; j >>= 1) {
      for (int i = threadIdx.x; i < BINCAP; i += 512) {
        int ixj = i ^ j;
        if (ixj > i) {
          bool up = ((i & k) == 0);
          bool aFirst = (s[i] > s[ixj]) || (s[i] == s[ixj] && id[i] < id[ixj]);
          bool doSwap = up ? !aFirst : aFirst;
          if (doSwap) {
            float ts = s[i]; s[i] = s[ixj]; s[ixj] = ts;
            int   ti = id[i]; id[i] = id[ixj]; id[ixj] = ti;
          }
        }
      }
      __syncthreads();
    }
  }
  int k1 = candCount[bc]; if (k1 > KSEL) k1 = KSEL;
  int need = cut[bc * 2 + 1];
  for (int t = threadIdx.x; t < KSEL - k1; t += 512) {
    float vs; int vi;
    if (t < need && t < cnt) { vs = s[t]; vi = id[t]; }
    else                     { vs = NEGV; vi = 0; }
    candS[bc * KSEL + k1 + t] = vs;
    candI[bc * KSEL + k1 + t] = vi;
  }
}

// ---------------- pass 5: per-(b,c) NMS over 512 candidates ----------------
__global__ void __launch_bounds__(512) det_nms_kernel(
    const float* __restrict__ boxes, const float* __restrict__ candS,
    const int* __restrict__ candI, float* __restrict__ selS, float* __restrict__ selB) {
  const int bc = blockIdx.x;
  const int b  = bc / NCLS;
  const int tid = threadIdx.x;

  __shared__ float by1[KSEL], bx1[KSEL], by2[KSEL], bx2[KSEL];
  __shared__ float area[KSEL], sc[KSEL];
  __shared__ unsigned int supp[KSEL * 16];   // 512x512 bit suppression matrix (32 KB)
  __shared__ float redS[KSEL];
  __shared__ int   redI[KSEL];

  // candidate scores: async copy global->LDS (ASYNCcnt path)
#if defined(__gfx1250__)
  {
    const float* gp = candS + (size_t)bc * KSEL + tid;
    unsigned ldsAddr = (unsigned)(size_t)&sc[tid];
    asm volatile("global_load_async_to_lds_b32 %0, %1, off" :: "v"(ldsAddr), "v"(gp) : "memory");
  }
#else
  sc[tid] = candS[(size_t)bc * KSEL + tid];
#endif

  int gi = candI[(size_t)bc * KSEL + tid];
  const float* bp = boxes + ((size_t)b * NBOX + gi) * 4;
  float y1 = bp[0], x1 = bp[1], y2 = bp[2], x2 = bp[3];
  by1[tid] = y1; bx1[tid] = x1; by2[tid] = y2; bx2[tid] = x2;
  area[tid] = (y2 - y1) * (x2 - x1);
  for (int w = tid; w < KSEL * 16; w += 512) supp[w] = 0u;

#if defined(__gfx1250__)
  asm volatile("s_wait_asynccnt 0" ::: "memory");
#endif
  __syncthreads();

  // pairwise suppression matrix: 32x32 tiles of 16x16; WMMA supplies area_i+area_j
  const int wave = tid >> 5;
  const int lane = tid & 31;
  const int laneLo = lane & 15;
  const int hi = (lane >> 4) & 1;
  for (int t = wave; t < 1024; t += 16) {
    const int i0 = (t >> 5) * 16;
    const int j0 = (t & 31) * 16;
    v8f d = tile_pair_sums(area, i0, j0, laneLo, hi);
    const int j = j0 + laneLo;
    const float cy1 = by1[j], cx1 = bx1[j], cy2 = by2[j], cx2 = bx2[j];
    const unsigned wordIdx = (unsigned)j >> 5;
    const unsigned bit = 1u << (j & 31);
    #pragma unroll
    for (int v = 0; v < 8; ++v) {
      const int i = i0 + v + hi * 8;
      float yy1 = fmaxf(by1[i], cy1);
      float xx1 = fmaxf(bx1[i], cx1);
      float yy2 = fminf(by2[i], cy2);
      float xx2 = fminf(bx2[i], cx2);
      float inter = fmaxf(yy2 - yy1, 0.0f) * fmaxf(xx2 - xx1, 0.0f);
      float interS = inter * AREA_SCALE;
      // iou > 0.5  <=>  2*inter > (areaSum - inter) + eps   (division-free)
      if (2.0f * interS > d[v] - interS + 1e-8f * AREA_SCALE)
        atomicOr(&supp[i * 16 + wordIdx], bit);
    }
  }
  __syncthreads();

  // 100 sequential selections: argmax (tie -> lowest index) + bitmask suppression
  for (int t = 0; t < MAXDET; ++t) {
    redS[tid] = sc[tid]; redI[tid] = tid;
    __syncthreads();
    for (int off = 256; off > 0; off >>= 1) {
      if (tid < off) {
        float s2 = redS[tid + off]; int i2 = redI[tid + off];
        if (s2 > redS[tid] || (s2 == redS[tid] && i2 < redI[tid])) { redS[tid] = s2; redI[tid] = i2; }
      }
      __syncthreads();
    }
    const int js = redI[0];
    if (tid == 0) {
      selS[(size_t)bc * MAXDET + t] = redS[0];
      float* ob = &selB[((size_t)bc * MAXDET + t) * 4];
      ob[0] = by1[js]; ob[1] = bx1[js]; ob[2] = by2[js]; ob[3] = bx2[js];
    }
    bool supd = (supp[js * 16 + (tid >> 5)] & (1u << (tid & 31))) != 0;
    if (supd || tid == js) sc[tid] = NEGV;
    __syncthreads();
  }
}

// ---------------- pass 6: per-batch top-100 merge over C*100 = 9000 entries
#define MERGE_N 9000
#define MERGE_P 9216
__global__ void __launch_bounds__(1024) det_final_kernel(
    const float* __restrict__ selS, const float* __restrict__ selB,
    float* __restrict__ outBoxes, float* __restrict__ outScores,
    float* __restrict__ outCls, float* __restrict__ outNum) {
  const int b = blockIdx.x;
  __shared__ float fs[MERGE_P];
  __shared__ float redS[1024];
  __shared__ int   redI[1024];
  __shared__ int   nvalid;
  for (int i = threadIdx.x; i < MERGE_P; i += 1024)
    fs[i] = (i < MERGE_N) ? selS[(size_t)b * MERGE_N + i] : NEGV;
  if (threadIdx.x == 0) nvalid = 0;
  __syncthreads();
  for (int t = 0; t < MAXDET; ++t) {
    float bs = NEGV; int bi = 0x7fffffff;
    for (int i = threadIdx.x; i < MERGE_P; i += 1024) {
      float v = fs[i];
      if (v > bs || (v == bs && i < bi)) { bs = v; bi = i; }
    }
    redS[threadIdx.x] = bs; redI[threadIdx.x] = bi;
    __syncthreads();
    for (int off = 512; off > 0; off >>= 1) {
      if (threadIdx.x < off) {
        float s2 = redS[threadIdx.x + off]; int i2 = redI[threadIdx.x + off];
        if (s2 > redS[threadIdx.x] || (s2 == redS[threadIdx.x] && i2 < redI[threadIdx.x])) {
          redS[threadIdx.x] = s2; redI[threadIdx.x] = i2;
        }
      }
      __syncthreads();
    }
    if (threadIdx.x == 0) {
      const int kk = redI[0];
      const float ss = redS[0];
      const bool valid = (ss >= SCORE_THR);
      const int cls = kk / MAXDET;
      const float* sb = &selB[((size_t)b * MERGE_N + kk) * 4];
      float* ob = &outBoxes[(b * MAXDET + t) * 4];
      ob[0] = valid ? sb[0] : 0.0f;
      ob[1] = valid ? sb[1] : 0.0f;
      ob[2] = valid ? sb[2] : 0.0f;
      ob[3] = valid ? sb[3] : 0.0f;
      outScores[b * MAXDET + t] = valid ? ss : 0.0f;
      outCls[b * MAXDET + t] = valid ? (float)cls : 0.0f;  // class ids exactly representable in f32
      if (valid) nvalid++;
      fs[kk] = NEGV;
    }
    __syncthreads();
  }
  if (threadIdx.x == 0) outNum[b] = (float)nvalid;
}

// ---------------- host launcher ----------------
extern "C" void kernel_launch(void* const* d_in, const int* in_sizes, int n_in,
                              void* d_out, int out_size, void* d_ws, size_t ws_size,
                              hipStream_t stream) {
  (void)in_sizes; (void)n_in; (void)out_size; (void)ws_size;
  const float* boxes  = (const float*)d_in[0];  // (B, N, 1, 4)
  const float* scores = (const float*)d_in[1];  // (B, N, C)

  // workspace layout (~13.3 MB total)
  char* ws = (char*)d_ws;
  size_t off = 0;
  int*   hist      = (int*)(ws + off);   off += (size_t)NBC * NBINS * 4;   // 2.95 MB
  int*   candCount = (int*)(ws + off);   off += (size_t)NBC * 4;
  int*   binCount  = (int*)(ws + off);   off += (size_t)NBC * 4;
  const size_t zeroBytes = off;                                            // only these need zeroing
  int*   cut   = (int*)(ws + off);       off += (size_t)NBC * 2 * 4;
  float* candS = (float*)(ws + off);     off += (size_t)NBC * KSEL * 4;
  int*   candI = (int*)(ws + off);       off += (size_t)NBC * KSEL * 4;
  float* binS  = (float*)(ws + off);     off += (size_t)NBC * BINCAP * 4;
  int*   binI  = (int*)(ws + off);       off += (size_t)NBC * BINCAP * 4;
  float* selS  = (float*)(ws + off);     off += (size_t)NBC * MAXDET * 4;
  float* selB  = (float*)(ws + off);     off += (size_t)NBC * MAXDET * 4 * 4;

  hipMemsetAsync(d_ws, 0, zeroBytes, stream);   // graph-capturable

  dim3 gStream((NC_FLAT + HCHUNK - 1) / HCHUNK, BATCH);
  det_hist_kernel   <<<gStream, 256, 0, stream>>>(scores, hist);
  det_cut_kernel    <<<NBC,     256, 0, stream>>>(hist, cut);
  det_compact_kernel<<<gStream, 256, 0, stream>>>(scores, cut, candS, candI, candCount,
                                                  binS, binI, binCount);
  det_binselect_kernel<<<NBC,   512, 0, stream>>>(binS, binI, binCount, cut, candS, candI, candCount);
  det_nms_kernel    <<<NBC,     512, 0, stream>>>(boxes, candS, candI, selS, selB);

  float* outBoxes  = (float*)d_out;                       // 8*100*4
  float* outScores = outBoxes + BATCH * MAXDET * 4;       // 8*100
  float* outCls    = outScores + BATCH * MAXDET;          // 8*100
  float* outNum    = outCls + BATCH * MAXDET;             // 8
  det_final_kernel  <<<BATCH,  1024, 0, stream>>>(selS, selB, outBoxes, outScores, outCls, outNum);
}
